// DHSRNN_71382356459528
// MI455X (gfx1250) — matrix operand order repaired and checked by reading, not compiled, not run on hardware
//
#include <hip/hip_runtime.h>
#include <hip/hip_bf16.h>
#include <math.h>

typedef _Float16 f16;
typedef _Float16 v16h __attribute__((ext_vector_type(16)));
typedef _Float16 v8h  __attribute__((ext_vector_type(8)));
typedef float    v8f  __attribute__((ext_vector_type(8)));

#define B_    128
#define T_    250
#define IN_   700
#define HID_  1024
#define BR_   4
#define OUT_  20
#define K_    1724     // IN + HID
#define KP_   1728     // padded K (multiple of 32)
#define N1_   4096     // HID*BR
#define KT_   54       // KP_/32
#define NT_   256      // N1_/16
#define NBLK_ 64
#define TPB_  256
#define NWAVE_ (NBLK_*8)     // 512 waves == 512 32x32 macro-tiles
#define VTH_  0.5f

// ---------------- workspace layout (bytes, all 256-aligned) ----------------
// NOTE: state buffers are TRANSPOSED (batch m innermost) so the WMMA epilogue
// (fixed n per lane, 8 consecutive m per lane) vectorizes to b128 accesses.
#define WPACK_OFF  0u
#define WPACK_B    (4096u*1728u*2u)              // f16 fragment-packed W1*mask
#define KIN_OFF    (WPACK_OFF + WPACK_B)
#define KIN_B      (128u*1728u*2u)               // f16 [B][KP] concat(x_t, spk_old, pad)
#define SPK_OFF    (KIN_OFF + KIN_B)
#define SPK_B      (128u*1024u*2u)               // f16 [HID][B] new spikes (transposed)
#define MEM1_OFF   (SPK_OFF + SPK_B)
#define MEM1_B     (128u*1024u*4u)               // f32 [HID][B] (transposed)
#define DINP_OFF   (MEM1_OFF + MEM1_B)
#define DINP_B     (128u*4096u*4u)               // f32 [N1][B] (transposed)
#define MEM2_OFF   (DINP_OFF + DINP_B)
#define MEM2_B     (128u*20u*4u)
#define SPK2_OFF   (MEM2_OFF + MEM2_B)
#define SPK2_B     (128u*20u*4u)
#define A1_OFF     (SPK2_OFF + SPK2_B)
#define A1_B       (1024u*4u)
#define BETA_OFF   (A1_OFF + A1_B)
#define BETA_B     (4096u*4u)
#define BAR_OFF    (BETA_OFF + BETA_B)
#define BAR_B      256u
#define WS_NEEDED  (BAR_OFF + BAR_B)             // ~17.5 MB

__device__ __forceinline__ float sigmoidf_(float v) { return 1.f / (1.f + expf(-v)); }

// DPP quad_perm lane-xor adds (branches of one neuron live in an aligned quad)
__device__ __forceinline__ float quad_xor1(float v) {  // [1,0,3,2] == xor 1
  return __int_as_float(__builtin_amdgcn_mov_dpp(__float_as_int(v), 0xB1, 0xF, 0xF, true));
}
__device__ __forceinline__ float quad_xor2(float v) {  // [2,3,0,1] == xor 2
  return __int_as_float(__builtin_amdgcn_mov_dpp(__float_as_int(v), 0x4E, 0xF, 0xF, true));
}

// ---------------------------------------------------------------------------
// Prep 1: state init + sigmoid precompute. Launched 2048x256 (524288 threads).
// ---------------------------------------------------------------------------
__global__ void snn_init(const float* __restrict__ mem1_0, const float* __restrict__ mem2_0,
                         const float* __restrict__ tau_m1, const float* __restrict__ tau_n,
                         float* __restrict__ mem1_t, float* __restrict__ d_inp_t,
                         float* __restrict__ mem2, float* __restrict__ spk2,
                         float* __restrict__ alpha1, float* __restrict__ beta_n,
                         f16* __restrict__ kin, f16* __restrict__ spk_t,
                         unsigned* __restrict__ bar) {
  int i = blockIdx.x * blockDim.x + threadIdx.x;
  if (i < 128 * 4096) d_inp_t[i] = 0.f;
  if (i < 128 * 1024) {
    // transpose: mem1_t[h][b] = mem1_0[b][h]
    mem1_t[i] = mem1_0[(size_t)(i & 127) * HID_ + (i >> 7)];
    spk_t[i] = (f16)0.f;
  }
  if (i < 128 * KP_)  kin[i] = (f16)0.f;          // zeros pad cols [1724,1728) forever
  if (i < 128 * OUT_) { mem2[i] = mem2_0[i]; spk2[i] = 0.f; }
  if (i < HID_)       alpha1[i] = sigmoidf_(tau_m1[i]);
  if (i < N1_)        beta_n[i] = sigmoidf_(tau_n[i]);   // tau_n flat == n = i*BR+j
  if (i < 64)         bar[i] = 0u;
}

// ---------------------------------------------------------------------------
// Prep 2: pack W1*mask -> f16, B-fragment-native tile layout.
// Tile (kt,nt): 32(K) x 16(N). lane: N = nt*16 + (lane&15); half h = lane/16:
// q<8 -> K = kt*32 + 8h + q, q>=8 -> K = kt*32 + 16 + 8h + (q-8).
// ---------------------------------------------------------------------------
__global__ void snn_pack(const float* __restrict__ W1, const int* __restrict__ mask,
                         f16* __restrict__ wp) {
  int tid = blockIdx.x * blockDim.x + threadIdx.x;
  if (tid >= KT_ * NT_ * 32) return;
  int lane = tid & 31;
  int tile = tid >> 5;             // tile = kt*NT_ + nt
  int nt = tile & 255;
  int kt = tile >> 8;
  int n_g = nt * 16 + (lane & 15);
  int kb  = kt * 32 + ((lane & 16) ? 8 : 0);
  f16* dst = wp + ((size_t)tile * 32 + lane) * 16;
  const float* wr = W1 + (size_t)n_g * K_;
  const int*   mr = mask + (size_t)n_g * K_;
#pragma unroll
  for (int q = 0; q < 16; ++q) {
    int k = kb + q + ((q >= 8) ? 8 : 0);
    float v = 0.f;
    if (k < K_ && mr[k] != 0) v = wr[k];
    dst[q] = (f16)v;
  }
}

// ---------------------------------------------------------------------------
// Grid-wide barrier: monotonically increasing counter, all NBLK_ blocks resident.
// ---------------------------------------------------------------------------
__device__ __forceinline__ void grid_barrier(unsigned* bar, unsigned target) {
  __syncthreads();
  if (threadIdx.x == 0) {
    __threadfence();
    __hip_atomic_fetch_add(bar, 1u, __ATOMIC_ACQ_REL, __HIP_MEMORY_SCOPE_AGENT);
    while (__hip_atomic_load(bar, __ATOMIC_ACQUIRE, __HIP_MEMORY_SCOPE_AGENT) < target) {
      __builtin_amdgcn_s_sleep(2);
    }
  }
  __syncthreads();
}

// ---------------------------------------------------------------------------
// Fused SNN epilogue for one 16x16 f32 fragment.
// C/D layout: N = lane&15, M = r (lanes 0-15) / 8+r (lanes 16-31).
// Transposed state -> per-lane b128 vector accesses (8 consecutive m).
// Old spike is recomputed from old mem1 (spk == (mem1_old - VTH > 0)), except
// at t==0 where the reference defines spk1_0 == 0.
// ---------------------------------------------------------------------------
__device__ __forceinline__ void tile_epilogue(
    v8f acc, int n_g, int mrow, int lane, int t,
    const float* __restrict__ b1, const float* __restrict__ beta_n,
    const float* __restrict__ alpha1, float* __restrict__ d_inp_t,
    float* __restrict__ mem1_t, f16* __restrict__ spk_t) {
  float betav = beta_n[n_g];
  float b1v   = b1[n_g];
  int   i_g   = n_g >> 2;
  float a1    = alpha1[i_g];
  float* dptr = d_inp_t + (size_t)n_g * B_ + mrow;
  v8f dold = *(const v8f*)dptr;
  v8f dnew;
#pragma unroll
  for (int r = 0; r < 8; ++r)
    dnew[r] = betav * dold[r] + (1.f - betav) * (acc[r] + b1v);
  *(v8f*)dptr = dnew;
  // sum the 4 branches (aligned lane quad) -> l_in ; pure VALU via DPP
  v8f s;
#pragma unroll
  for (int r = 0; r < 8; ++r) {
    float sr = dnew[r] + quad_xor1(dnew[r]);
    s[r] = sr + quad_xor2(sr);
  }
  if ((lane & 3) == 0) {
    float* mptr = mem1_t + (size_t)i_g * B_ + mrow;
    v8f mold = *(const v8f*)mptr;
    v8f mnew;
    v8h sp;
#pragma unroll
    for (int r = 0; r < 8; ++r) {
      float sold = (t > 0 && (mold[r] - VTH_) > 0.f) ? 1.f : 0.f;
      float mn = mold[r] * a1 + (1.f - a1) * s[r] - VTH_ * sold;
      mnew[r] = mn;
      sp[r] = (f16)(((mn - VTH_) > 0.f) ? 1.f : 0.f);
    }
    *(v8f*)mptr = mnew;
    *(v8h*)(spk_t + (size_t)i_g * B_ + mrow) = sp;
  }
}

// ---------------------------------------------------------------------------
// Phase 2: d2 = spk @ W2^T, mem2/spk2 update, log_softmax, out[:, :, t].
// Block bid owns batch rows 2*bid, 2*bid+1.
// ---------------------------------------------------------------------------
__device__ __forceinline__ void phase2_fn(
    int t, int tid, int lane, int bid,
    const f16* __restrict__ spk_t, const float* __restrict__ W2,
    const float* __restrict__ b2, const float* __restrict__ tau_m2,
    float* __restrict__ mem2, float* __restrict__ spk2, float* __restrict__ out,
    float (*d2s)[20], float (*m2s)[24]) {
  int wv = tid >> 5;
#pragma unroll 1
  for (int d = 0; d < 5; ++d) {
    int idx = wv * 5 + d;               // 0..39
    int b_l = idx / OUT_, o = idx % OUT_;
    int b = bid * 2 + b_l;
    const float* w2r = W2 + (size_t)o * HID_;
    float dot = 0.f;
#pragma unroll 4
    for (int c = 0; c < HID_ / 32; ++c) {
      int h = lane + c * 32;
      dot += (float)spk_t[(size_t)h * B_ + b] * w2r[h];
    }
    for (int off = 16; off > 0; off >>= 1) dot += __shfl_xor(dot, off, 32);
    if (lane == 0) d2s[b_l][o] = dot;
  }
  __syncthreads();
  if (tid < 2 * OUT_) {
    int b_l = tid / OUT_, o = tid % OUT_;
    int b = bid * 2 + b_l;
    int mi = b * OUT_ + o;
    float a2 = sigmoidf_(tau_m2[o]);
    float d2v = d2s[b_l][o] + b2[o];
    float m2 = mem2[mi] * a2 + (1.f - a2) * d2v - VTH_ * spk2[mi];
    spk2[mi] = (m2 - VTH_) > 0.f ? 1.f : 0.f;
    mem2[mi] = m2;
    m2s[b_l][o] = m2;
  }
  __syncthreads();
  if (tid < 2) {
    int b = bid * 2 + tid;
    float mx = m2s[tid][0];
    for (int o = 1; o < OUT_; ++o) mx = fmaxf(mx, m2s[tid][o]);
    float se = 0.f;
    for (int o = 0; o < OUT_; ++o) se += expf(m2s[tid][o] - mx);
    float ls = mx + logf(se);
    for (int o = 0; o < OUT_; ++o)
      out[((size_t)b * OUT_ + o) * T_ + t] = m2s[tid][o] - ls;
  }
  __syncthreads();   // protect d2s/m2s reuse
}

// ---------------------------------------------------------------------------
// Persistent kernel: whole T=250 recurrence. 64 blocks x 256 threads.
// Per step: {P2(t-1); P0(t)} -> bar -> P1(t) -> bar   (2 grid barriers/step)
// ---------------------------------------------------------------------------
__global__ __launch_bounds__(TPB_) void snn_main(
    const float* __restrict__ x, const float* __restrict__ b1,
    const float* __restrict__ W2, const float* __restrict__ b2,
    const float* __restrict__ tau_m2, float* __restrict__ out,
    const f16* __restrict__ wp, f16* __restrict__ kin, f16* __restrict__ spk_t,
    float* __restrict__ mem1_t, float* __restrict__ d_inp_t,
    float* __restrict__ mem2, float* __restrict__ spk2,
    const float* __restrict__ alpha1, const float* __restrict__ beta_n,
    unsigned* __restrict__ bar) {
  const int tid  = threadIdx.x;
  const int lane = tid & 31;
  const int bid  = blockIdx.x;
  const int gwave = bid * 8 + (tid >> 5);
  const int gthreads = NBLK_ * TPB_;
  const int gtid = bid * TPB_ + tid;
  // macro-tile: rows [mt2*32, +32), cols [nt2*32, +32)
  const int mt2 = gwave >> 7;      // 0..3
  const int nt2 = gwave & 127;     // 0..127
  __shared__ float d2s[2][20];
  __shared__ float m2s[2][24];
  unsigned gen = 0;

#pragma unroll 1
  for (int t = 0; t < T_; ++t) {
    // ---------- Phase 2 of previous step (independent of Phase 0) ----------
    if (t > 0)
      phase2_fn(t - 1, tid, lane, bid, spk_t, W2, b2, tau_m2, mem2, spk2, out, d2s, m2s);

    // ---------- Phase 0: kin = concat(x_t, spk_old) in f16 ----------
    for (int idx = gtid; idx < B_ * K_; idx += gthreads) {
      int b = idx / K_;
      int k = idx - b * K_;
      f16 v;
      if (k < IN_) v = (f16)x[((size_t)b * T_ + t) * IN_ + k];
      else         v = spk_t[(size_t)(k - IN_) * B_ + b];
      kin[(size_t)b * KP_ + k] = v;
    }
    ++gen; grid_barrier(bar, gen * NBLK_);

    // ---------- Phase 1: ff = kin @ W1m^T (WMMA f16->f32), fused SNN update ----------
    {
      // A-fragment addressing (16-bit A 16x32 layout):
      // lanes 0-15: K q=0..7 -> k, q=8..15 -> k+16 ; lanes 16-31: +8
      const int kbaseA = (lane & 16) ? 8 : 0;
      const f16* arow0 = kin + (size_t)(mt2 * 32 + (lane & 15)) * KP_;
      const f16* arow1 = arow0 + (size_t)16 * KP_;
      const f16* bbase0 = wp + ((size_t)(nt2 * 2) * 32 + lane) * 16;       // n-tile 2*nt2
      const f16* bbase1 = wp + ((size_t)(nt2 * 2 + 1) * 32 + lane) * 16;   // n-tile 2*nt2+1
      v8f acc00 = {0.f,0.f,0.f,0.f,0.f,0.f,0.f,0.f};
      v8f acc01 = acc00, acc10 = acc00, acc11 = acc00;
#pragma unroll 2
      for (int kt = 0; kt < KT_; ++kt) {
        const size_t ktile = (size_t)kt * NT_ * 32 * 16;   // halves per kt row of tiles
        v8h a0lo = *(const v8h*)(arow0 + kt * 32 + kbaseA);
        v8h a0hi = *(const v8h*)(arow0 + kt * 32 + kbaseA + 16);
        v8h a1lo = *(const v8h*)(arow1 + kt * 32 + kbaseA);
        v8h a1hi = *(const v8h*)(arow1 + kt * 32 + kbaseA + 16);
        v16h a0 = __builtin_shufflevector(a0lo, a0hi, 0,1,2,3,4,5,6,7,8,9,10,11,12,13,14,15);
        v16h a1 = __builtin_shufflevector(a1lo, a1hi, 0,1,2,3,4,5,6,7,8,9,10,11,12,13,14,15);
        v16h bf0 = *(const v16h*)(bbase0 + ktile);
        v16h bf1 = *(const v16h*)(bbase1 + ktile);
        acc00 = __builtin_amdgcn_wmma_f32_16x16x32_f16(false, a0, false, bf0, (short)0, acc00, false, false);
        acc01 = __builtin_amdgcn_wmma_f32_16x16x32_f16(false, a0, false, bf1, (short)0, acc01, false, false);
        acc10 = __builtin_amdgcn_wmma_f32_16x16x32_f16(false, a1, false, bf0, (short)0, acc10, false, false);
        acc11 = __builtin_amdgcn_wmma_f32_16x16x32_f16(false, a1, false, bf1, (short)0, acc11, false, false);
      }
      const int nlo  = nt2 * 32 + (lane & 15);
      const int nhi  = nlo + 16;
      const int mlo  = mt2 * 32 + ((lane & 16) ? 8 : 0);
      const int mhi  = mlo + 16;
      tile_epilogue(acc00, nlo, mlo, lane, t, b1, beta_n, alpha1, d_inp_t, mem1_t, spk_t);
      tile_epilogue(acc01, nhi, mlo, lane, t, b1, beta_n, alpha1, d_inp_t, mem1_t, spk_t);
      tile_epilogue(acc10, nlo, mhi, lane, t, b1, beta_n, alpha1, d_inp_t, mem1_t, spk_t);
      tile_epilogue(acc11, nhi, mhi, lane, t, b1, beta_n, alpha1, d_inp_t, mem1_t, spk_t);
    }
    ++gen; grid_barrier(bar, gen * NBLK_);
  }
  // ---------- Final Phase 2 for t = T-1 ----------
  phase2_fn(T_ - 1, tid, lane, bid, spk_t, W2, b2, tau_m2, mem2, spk2, out, d2s, m2s);
}

// ---------------------------------------------------------------------------
extern "C" void kernel_launch(void* const* d_in, const int* in_sizes, int n_in,
                              void* d_out, int out_size, void* d_ws, size_t ws_size,
                              hipStream_t stream) {
  (void)in_sizes; (void)n_in; (void)out_size;
  if (ws_size < (size_t)WS_NEEDED) return;

  const float* x      = (const float*)d_in[0];
  const float* W1     = (const float*)d_in[1];
  const float* b1     = (const float*)d_in[2];
  const float* tau_m1 = (const float*)d_in[3];
  const float* tau_n  = (const float*)d_in[4];
  const float* W2     = (const float*)d_in[5];
  const float* b2     = (const float*)d_in[6];
  const float* tau_m2 = (const float*)d_in[7];
  const float* mem1_0 = (const float*)d_in[8];
  const float* mem2_0 = (const float*)d_in[9];
  const int*   mask   = (const int*)d_in[10];
  float* out = (float*)d_out;

  char* ws = (char*)d_ws;
  f16*      wpack   = (f16*)(ws + WPACK_OFF);
  f16*      kin     = (f16*)(ws + KIN_OFF);
  f16*      spk_t   = (f16*)(ws + SPK_OFF);
  float*    mem1_t  = (float*)(ws + MEM1_OFF);
  float*    d_inp_t = (float*)(ws + DINP_OFF);
  float*    mem2    = (float*)(ws + MEM2_OFF);
  float*    spk2    = (float*)(ws + SPK2_OFF);
  float*    alpha1  = (float*)(ws + A1_OFF);
  float*    beta_n  = (float*)(ws + BETA_OFF);
  unsigned* bar     = (unsigned*)(ws + BAR_OFF);

  snn_init<<<2048, 256, 0, stream>>>(mem1_0, mem2_0, tau_m1, tau_n,
                                     mem1_t, d_inp_t, mem2, spk2, alpha1, beta_n,
                                     kin, spk_t, bar);
  snn_pack<<<1728, 256, 0, stream>>>(W1, mask, wpack);
  snn_main<<<NBLK_, TPB_, 0, stream>>>(x, b1, W2, b2, tau_m2, out,
                                       wpack, kin, spk_t, mem1_t, d_inp_t,
                                       mem2, spk2, alpha1, beta_n, bar);
}
